// GatedDeltaNetAttention_66984309948669
// MI455X (gfx1250) — compile-verified, matrix operand order checked by reading, not run on hardware
//
#include <hip/hip_runtime.h>
#include <hip/hip_bf16.h>

// ---------------------------------------------------------------------------
// Gated linear attention (GatedDeltaNet-style, scalar per-head decay) for
// MI455X / gfx1250.  All heavy math on bf16 WMMA (v_wmma_f32_16x16x32_bf16,
// f32 accumulate).  Scan is chunked (chunk=32) so the sequential part is also
// WMMA GEMMs; 128x128 f32 state lives in LDS per workgroup.  V-chunk staging
// uses GLOBAL_LOAD_ASYNC_TO_LDS_B128 (ASYNCcnt).  Big GEMMs stream operands
// from global (whole working set < 140MB fits the 192MB L2) with a software-
// pipelined (ping-pong) K loop + global_prefetch.
// ---------------------------------------------------------------------------

constexpr int Hdim = 2048;   // hidden
constexpr int SEQ  = 2048;
constexpr int HK   = 16;     // key heads
constexpr int KDIM = 128;    // key dim
constexpr int HV   = 32;
constexpr int DVd  = 128;
constexpr int VEF  = 256;    // (HV/HK)*DV, per key head
constexpr int VD   = HV * DVd;  // 4096
constexpr int CKW  = 4;      // conv kernel width
constexpr int CH   = 32;     // scan chunk

// padded LDS strides (bank-conflict avoidance for column-strided gathers)
constexpr int VSTR = 136;    // q/k/v chunk row stride (ushorts)
constexpr int SSTR = 132;    // state row stride (floats)
constexpr int ASTR = 40;     // attn tile row stride (ushorts)

typedef __attribute__((ext_vector_type(16))) __bf16         bf16x16;
typedef __attribute__((ext_vector_type(8)))  float          f32x8;
typedef __attribute__((ext_vector_type(8)))  unsigned short u16x8;

union Frag { bf16x16 v; unsigned short u[16]; };

__device__ __forceinline__ unsigned short f2bf(float f) {
  unsigned int u = __builtin_bit_cast(unsigned int, f);
  u += 0x7FFFu + ((u >> 16) & 1u);          // round-to-nearest-even
  return (unsigned short)(u >> 16);
}
__device__ __forceinline__ float bf2f(unsigned short h) {
  unsigned int u = ((unsigned int)h) << 16;
  return __builtin_bit_cast(float, u);
}
__device__ __forceinline__ f32x8 zero8() {
  f32x8 z;
#pragma unroll
  for (int i = 0; i < 8; ++i) z[i] = 0.f;
  return z;
}

// WMMA 16x16x32 bf16 fragment lane striping (wave32):
//  lanes 0-15 : element index within lane covers K {k0..k0+7, k0+16..k0+23}
//  lanes 16-31: K {k0+8..k0+15, k0+24..k0+31}
// A operand: lane holds row (rowbase + lane&15).  B operand: lane holds
// column (colbase + lane&15); since C = A @ W^T, a row of W is a column of B,
// so both operands load with the same K-contiguous pattern.
__device__ __forceinline__ Frag ldg_frag(const unsigned short* base, int ld,
                                         int rowbase, int k0, int lane) {
  Frag f;
  const unsigned short* p =
      base + (size_t)(rowbase + (lane & 15)) * ld + k0 + ((lane >> 4) << 3);
  u16x8 lo = *(const u16x8*)p;
  u16x8 hi = *(const u16x8*)(p + 16);
#pragma unroll
  for (int i = 0; i < 8; ++i) { f.u[i] = lo[i]; f.u[8 + i] = hi[i]; }
  return f;
}

__device__ __forceinline__ Frag lds_rowfrag(const unsigned short* buf, int ld,
                                            int rowbase, int k0, int lane) {
  Frag f;
  const unsigned short* p =
      buf + (rowbase + (lane & 15)) * ld + k0 + ((lane >> 4) << 3);
#pragma unroll
  for (int i = 0; i < 8; ++i) { f.u[i] = p[i]; f.u[8 + i] = p[16 + i]; }
  return f;
}

// B fragment when the K dimension strides through LDS rows (buf is [k][ld]).
__device__ __forceinline__ Frag lds_colfrag(const unsigned short* buf, int ld,
                                            int k0, int colbase, int lane) {
  Frag f;
  int col = colbase + (lane & 15), kh = (lane >> 4) << 3;
#pragma unroll
  for (int i = 0; i < 8; ++i) {
    f.u[i]     = buf[(k0 + kh + i) * ld + col];
    f.u[8 + i] = buf[(k0 + 16 + kh + i) * ld + col];
  }
  return f;
}

__device__ __forceinline__ Frag lds_colfrag_f32(const float* buf, int ld,
                                                int k0, int colbase, int lane) {
  Frag f;
  int col = colbase + (lane & 15), kh = (lane >> 4) << 3;
#pragma unroll
  for (int i = 0; i < 8; ++i) {
    f.u[i]     = f2bf(buf[(k0 + kh + i) * ld + col]);
    f.u[8 + i] = f2bf(buf[(k0 + 16 + kh + i) * ld + col]);
  }
  return f;
}

// ---------------------------------------------------------------------------
// elementwise f32 -> bf16 conversion
// ---------------------------------------------------------------------------
__global__ void cvt_f32_bf16(const float* __restrict__ s,
                             unsigned short* __restrict__ d, int n) {
  int i = blockIdx.x * blockDim.x + threadIdx.x;
  if (i < n) d[i] = f2bf(s[i]);
}

// torch Conv1d weight (out=2048, in/groups=128, k=4) -> bf16 [tau][c][j]
__global__ void repack_conv(const float* __restrict__ w,
                            unsigned short* __restrict__ d, int n) {
  int i = blockIdx.x * blockDim.x + threadIdx.x;
  if (i >= n) return;
  int c = i >> 9;           // / (128*4)
  int rem = i & 511;
  int j = rem >> 2;
  int tau = rem & 3;
  d[((size_t)tau * Hdim + c) * KDIM + j] = f2bf(w[i]);
}

// ---------------------------------------------------------------------------
// C[M,N] = A[M,K] @ W[N,K]^T  (bf16 inputs, f32 accum).  Tile 128x128 per
// block (8 waves as 4M x 2N, 32x64 per wave).  K loop is software pipelined:
// ping-pong fragment sets so loads for step n+1 issue before WMMAs of step n.
// ---------------------------------------------------------------------------
template <bool BF16OUT>
__global__ __launch_bounds__(256) void gemm_nt(
    const unsigned short* __restrict__ A, const unsigned short* __restrict__ W,
    void* __restrict__ C, int M, int N, int K) {
  int lane = threadIdx.x & 31, wave = threadIdx.x >> 5;
  int mbase = blockIdx.y * 128 + (wave & 3) * 32;
  int nbase = blockIdx.x * 128 + (wave >> 2) * 64;
  f32x8 acc[2][4];
#pragma unroll
  for (int i = 0; i < 2; ++i)
#pragma unroll
    for (int j = 0; j < 4; ++j) acc[i][j] = zero8();

  Frag aP[2], bP[4], aQ[2], bQ[4];
#pragma unroll
  for (int mt = 0; mt < 2; ++mt) aP[mt] = ldg_frag(A, K, mbase + mt * 16, 0, lane);
#pragma unroll
  for (int nt = 0; nt < 4; ++nt) bP[nt] = ldg_frag(W, K, nbase + nt * 16, 0, lane);

  for (int k0 = 0; k0 < K; k0 += 64) {
    // issue loads for the odd half-step before consuming the even one
#pragma unroll
    for (int mt = 0; mt < 2; ++mt) aQ[mt] = ldg_frag(A, K, mbase + mt * 16, k0 + 32, lane);
#pragma unroll
    for (int nt = 0; nt < 4; ++nt) bQ[nt] = ldg_frag(W, K, nbase + nt * 16, k0 + 32, lane);
    if (k0 + 256 < K) {  // pull L2 lines toward the WGP ahead of time
      __builtin_prefetch(A + (size_t)(mbase + (lane & 15)) * K + k0 + 256, 0, 0);
      __builtin_prefetch(W + (size_t)(nbase + (lane & 15)) * K + k0 + 256, 0, 0);
    }
#pragma unroll
    for (int mt = 0; mt < 2; ++mt)
#pragma unroll
      for (int nt = 0; nt < 4; ++nt)
        acc[mt][nt] = __builtin_amdgcn_wmma_f32_16x16x32_bf16(
            false, aP[mt].v, false, bP[nt].v, (short)0, acc[mt][nt], false, false);
    if (k0 + 64 < K) {
#pragma unroll
      for (int mt = 0; mt < 2; ++mt) aP[mt] = ldg_frag(A, K, mbase + mt * 16, k0 + 64, lane);
#pragma unroll
      for (int nt = 0; nt < 4; ++nt) bP[nt] = ldg_frag(W, K, nbase + nt * 16, k0 + 64, lane);
    }
#pragma unroll
    for (int mt = 0; mt < 2; ++mt)
#pragma unroll
      for (int nt = 0; nt < 4; ++nt)
        acc[mt][nt] = __builtin_amdgcn_wmma_f32_16x16x32_bf16(
            false, aQ[mt].v, false, bQ[nt].v, (short)0, acc[mt][nt], false, false);
  }

  int hi = (lane >> 4) << 3;
#pragma unroll
  for (int mt = 0; mt < 2; ++mt)
#pragma unroll
    for (int nt = 0; nt < 4; ++nt)
#pragma unroll
      for (int r = 0; r < 8; ++r) {
        int row = mbase + mt * 16 + r + hi;
        int col = nbase + nt * 16 + (lane & 15);
        float v = acc[mt][nt][r];
        if constexpr (BF16OUT)
          ((unsigned short*)C)[(size_t)row * N + col] = f2bf(v);
        else
          ((float*)C)[(size_t)row * N + col] = v;
      }
}

// ---------------------------------------------------------------------------
// Grouped causal conv (groups=16, K=4) + bias + SiLU as 4 shifted WMMA GEMMs
// with K=128 (the group's in-channels).  One wave = one 16x16 (t x c) tile.
// The causal-boundary guard is wave-uniform: only wave 0 of t-block 0 can see
// t-3+tau < 0, every other wave takes the branch-free interior path.
// ---------------------------------------------------------------------------
__global__ __launch_bounds__(256) void conv_silu(
    const unsigned short* __restrict__ xr,     // (SEQ, 2048) bf16 raw proj
    const unsigned short* __restrict__ wrep,   // [4][2048][128] bf16
    const float* __restrict__ bias,            // (2048,)
    unsigned short* __restrict__ y) {          // (SEQ, 2048) bf16
  int lane = threadIdx.x & 31, wave = threadIdx.x >> 5;
  int tb = blockIdx.y * 128 + wave * 16;
  int cb = blockIdx.x * 16;
  int g = cb >> 7;  // group
  f32x8 acc = zero8();
  if (tb >= CKW - 1) {  // interior: no causal boundary in this 16-row tile
#pragma unroll
    for (int tau = 0; tau < CKW; ++tau)
#pragma unroll
      for (int kt = 0; kt < 4; ++kt) {
        Frag a = ldg_frag(xr, Hdim, tb - (CKW - 1) + tau, g * KDIM + kt * 32, lane);
        Frag b = ldg_frag(wrep + (size_t)tau * Hdim * KDIM, KDIM, cb, kt * 32, lane);
        acc = __builtin_amdgcn_wmma_f32_16x16x32_bf16(false, a.v, false, b.v,
                                                      (short)0, acc, false, false);
      }
  } else {
#pragma unroll
    for (int tau = 0; tau < CKW; ++tau)
#pragma unroll
      for (int kt = 0; kt < 4; ++kt) {
        Frag a;
        int r = tb + (lane & 15) - (CKW - 1) + tau;
        if (r < 0) {
#pragma unroll
          for (int i = 0; i < 16; ++i) a.u[i] = 0;
        } else {
          const unsigned short* p =
              xr + (size_t)r * Hdim + g * KDIM + kt * 32 + ((lane >> 4) << 3);
          u16x8 lo = *(const u16x8*)p;
          u16x8 hiu = *(const u16x8*)(p + 16);
#pragma unroll
          for (int i = 0; i < 8; ++i) { a.u[i] = lo[i]; a.u[8 + i] = hiu[i]; }
        }
        Frag b = ldg_frag(wrep + (size_t)tau * Hdim * KDIM, KDIM, cb, kt * 32, lane);
        acc = __builtin_amdgcn_wmma_f32_16x16x32_bf16(false, a.v, false, b.v,
                                                      (short)0, acc, false, false);
      }
  }
  int hi = (lane >> 4) << 3;
#pragma unroll
  for (int r = 0; r < 8; ++r) {
    int t = tb + r + hi;
    int c = cb + (lane & 15);
    float v = acc[r] + bias[c];
    v = v / (1.f + __expf(-v));  // silu
    y[(size_t)t * Hdim + c] = f2bf(v);
  }
}

// ---------------------------------------------------------------------------
// gate / beta: sigmoid(x @ W^T + b), 16 outputs each per row (tiny N).
// ---------------------------------------------------------------------------
__global__ __launch_bounds__(256) void gate_beta_kernel(
    const float* __restrict__ x, const float* __restrict__ Wg,
    const float* __restrict__ bg, const float* __restrict__ Wb,
    const float* __restrict__ bb, float* __restrict__ gate,
    float* __restrict__ beta) {
  __shared__ float xrow[Hdim];
  __shared__ float part[32][8];
  int row = blockIdx.x;
  for (int i = threadIdx.x; i < Hdim; i += 256) xrow[i] = x[(size_t)row * Hdim + i];
  __syncthreads();
  int out = threadIdx.x >> 3;  // 0..31 (16 gate + 16 beta)
  int seg = threadIdx.x & 7;
  const float* Wrow = (out < 16) ? (Wg + (size_t)out * Hdim)
                                 : (Wb + (size_t)(out - 16) * Hdim);
  float s = 0.f;
  for (int j = seg * 256; j < seg * 256 + 256; ++j) s += xrow[j] * Wrow[j];
  part[out][seg] = s;
  __syncthreads();
  if (threadIdx.x < 32) {
    int o = threadIdx.x;
    float t = 0.f;
#pragma unroll
    for (int i = 0; i < 8; ++i) t += part[o][i];
    if (o < 16) {
      t += bg[o];
      gate[(size_t)row * HK + o] = 1.f / (1.f + __expf(-t));
    } else {
      int h = o - 16;
      t += bb[h];
      beta[(size_t)row * HK + h] = 1.f / (1.f + __expf(-t));
    }
  }
}

// ---------------------------------------------------------------------------
// Chunked scan.  32 blocks = (head, half of VEFF).  128x128 f32 state in LDS.
// Per chunk (C=32), with L_t = sum_{t'<=t} log b_t' (within chunk):
//   q~_t = q_t e^{L_t},  k^_s = k_s e^{-L_s}
//   o_t  = g_t * ( q~_t @ S_old  +  sum_{s<=t} (q~_t . k^_s) v_s )
//   S    = e^{L_last} * ( S_old + k^^T @ v )
// All four matrix ops are bf16 WMMAs; decays in log space.  V staging uses
// GLOBAL_LOAD_ASYNC_TO_LDS_B128 (inst offset applies to both sides per ISA).
// ---------------------------------------------------------------------------
__global__ __launch_bounds__(256) void scan_kernel(
    const unsigned short* __restrict__ qb,  // (SEQ, 2048) bf16 post conv+silu
    const unsigned short* __restrict__ kb,
    const unsigned short* __restrict__ vb,  // (SEQ, 4096) bf16
    const float* __restrict__ gate,         // (SEQ, 16)
    const float* __restrict__ beta,         // (SEQ, 16)
    unsigned short* __restrict__ attn_out,  // (SEQ, 4096) bf16
    float* __restrict__ Sf) {               // (16, 128, 256) f32
  extern __shared__ char smem[];
  float* S = (float*)smem;                                   // 128 x SSTR f32
  unsigned short* qs = (unsigned short*)(smem + KDIM * SSTR * 4);
  unsigned short* ks = qs + CH * VSTR;
  unsigned short* vs = ks + CH * VSTR;
  unsigned short* attnA = vs + CH * VSTR;                    // 32 x ASTR bf16
  float* Lpre = (float*)(attnA + CH * ASTR);
  float* gch = Lpre + CH;
  float* lbuf = gch + CH;

  int head = blockIdx.x >> 1, half = blockIdx.x & 1;
  int tid = threadIdx.x, lane = tid & 31, wave = tid >> 5;

  for (int i = tid; i < KDIM * SSTR; i += 256) S[i] = 0.f;
  __syncthreads();

  for (int c = 0; c < SEQ / CH; ++c) {
    int t0 = c * CH;
    if (tid < CH) {
      lbuf[tid] = __logf(fmaxf(beta[(size_t)(t0 + tid) * HK + head], 1e-30f));
      gch[tid] = gate[(size_t)(t0 + tid) * HK + head];
    }
    __syncthreads();
    if (tid < CH) {
      float s = 0.f;
      for (int i = 0; i <= tid; ++i) s += lbuf[i];
      Lpre[tid] = s;
    }
    __syncthreads();
    {  // stage: scaled q~,k^ via VALU; raw v via async global->LDS DMA
      int row = tid >> 3, d0 = (tid & 7) * 16;
      const unsigned short* vg =
          vb + (size_t)(t0 + row) * VD + head * VEF + half * KDIM + d0;
      unsigned ldsv = (unsigned)(uintptr_t)(vs + row * VSTR + d0);
      asm volatile("global_load_async_to_lds_b128 %0, %1, off"
                   :: "v"(ldsv), "v"(vg) : "memory");
      asm volatile("global_load_async_to_lds_b128 %0, %1, off offset:16"
                   :: "v"(ldsv), "v"(vg) : "memory");
      float eq = __expf(Lpre[row]);
      float ek = __expf(-Lpre[row]);
      const unsigned short* qg = qb + (size_t)(t0 + row) * Hdim + head * KDIM + d0;
      const unsigned short* kg = kb + (size_t)(t0 + row) * Hdim + head * KDIM + d0;
#pragma unroll
      for (int i = 0; i < 16; ++i) {
        qs[row * VSTR + d0 + i] = f2bf(bf2f(qg[i]) * eq);
        ks[row * VSTR + d0 + i] = f2bf(bf2f(kg[i]) * ek);
      }
      asm volatile("s_wait_asynccnt 0x0" ::: "memory");
    }
    __syncthreads();

    int nw = wave * 16;  // each wave owns 16 of 128 output columns
    f32x8 o0 = zero8(), o1 = zero8();
    // inter-chunk: q~ @ S_old  (K = 128)
#pragma unroll
    for (int kt = 0; kt < 4; ++kt) {
      Frag bS = lds_colfrag_f32(S, SSTR, kt * 32, nw, lane);
      Frag a0 = lds_rowfrag(qs, VSTR, 0, kt * 32, lane);
      Frag a1 = lds_rowfrag(qs, VSTR, 16, kt * 32, lane);
      o0 = __builtin_amdgcn_wmma_f32_16x16x32_bf16(false, a0.v, false, bS.v,
                                                   (short)0, o0, false, false);
      o1 = __builtin_amdgcn_wmma_f32_16x16x32_bf16(false, a1.v, false, bS.v,
                                                   (short)0, o1, false, false);
    }
    // intra-chunk attention matrix (waves 0-3, one 16x16 tile each) + mask
    if (wave < 4) {
      int mt = wave >> 1, st = wave & 1;
      f32x8 a = zero8();
#pragma unroll
      for (int kt = 0; kt < 4; ++kt) {
        Frag aq = lds_rowfrag(qs, VSTR, mt * 16, kt * 32, lane);
        Frag bk = lds_rowfrag(ks, VSTR, st * 16, kt * 32, lane);
        a = __builtin_amdgcn_wmma_f32_16x16x32_bf16(false, aq.v, false, bk.v,
                                                    (short)0, a, false, false);
      }
      int hi = (lane >> 4) << 3;
#pragma unroll
      for (int r = 0; r < 8; ++r) {
        int tr = mt * 16 + r + hi, sc = st * 16 + (lane & 15);
        attnA[tr * ASTR + sc] = f2bf((sc <= tr) ? a[r] : 0.f);
      }
    }
    __syncthreads();
    // intra-chunk: A @ v  (K = 32, one WMMA per M-tile)
    {
      Frag bv = lds_colfrag(vs, VSTR, 0, nw, lane);
      Frag a0 = lds_rowfrag(attnA, ASTR, 0, 0, lane);
      Frag a1 = lds_rowfrag(attnA, ASTR, 16, 0, lane);
      o0 = __builtin_amdgcn_wmma_f32_16x16x32_bf16(false, a0.v, false, bv.v,
                                                   (short)0, o0, false, false);
      o1 = __builtin_amdgcn_wmma_f32_16x16x32_bf16(false, a1.v, false, bv.v,
                                                   (short)0, o1, false, false);
    }
    // apply output gate, store
    {
      int hi = (lane >> 4) << 3;
#pragma unroll
      for (int r = 0; r < 8; ++r) {
        int tr0 = r + hi, tr1 = 16 + r + hi;
        int col = head * VEF + half * KDIM + nw + (lane & 15);
        attn_out[(size_t)(t0 + tr0) * VD + col] = f2bf(o0[r] * gch[tr0]);
        attn_out[(size_t)(t0 + tr1) * VD + col] = f2bf(o1[r] * gch[tr1]);
      }
    }
    // state update: S = e^{L_last} (S + k^^T @ v); wave owns 16 d-rows
    {
      float eL = __expf(Lpre[CH - 1]);
      Frag aK = lds_colfrag(ks, VSTR, 0, wave * 16, lane);  // A[d][s] = k^[s][d]
      int hi = (lane >> 4) << 3;
#pragma unroll
      for (int nt = 0; nt < 8; ++nt) {
        Frag bv = lds_colfrag(vs, VSTR, 0, nt * 16, lane);
        f32x8 ds = __builtin_amdgcn_wmma_f32_16x16x32_bf16(
            false, aK.v, false, bv.v, (short)0, zero8(), false, false);
#pragma unroll
        for (int r = 0; r < 8; ++r) {
          int d = wave * 16 + r + hi, n = nt * 16 + (lane & 15);
          int idx = d * SSTR + n;
          S[idx] = eL * (S[idx] + ds[r]);
        }
      }
    }
    __syncthreads();
  }
  // final state -> d_out tail  (HK, KD, VEFF)
  for (int i = tid; i < KDIM * KDIM; i += 256) {
    int d = i >> 7, n = i & 127;
    Sf[((size_t)head * KDIM + d) * VEF + half * KDIM + n] = S[d * SSTR + n];
  }
}

// ---------------------------------------------------------------------------
extern "C" void kernel_launch(void* const* d_in, const int* in_sizes, int n_in,
                              void* d_out, int out_size, void* d_ws,
                              size_t ws_size, hipStream_t stream) {
  (void)in_sizes; (void)n_in; (void)out_size; (void)ws_size;
  const float* x   = (const float*)d_in[0];
  const float* Wq  = (const float*)d_in[1];
  const float* Wk  = (const float*)d_in[2];
  const float* Wv  = (const float*)d_in[3];
  const float* Wo  = (const float*)d_in[4];
  const float* Wqc = (const float*)d_in[5];
  const float* bqc = (const float*)d_in[6];
  const float* Wkc = (const float*)d_in[7];
  const float* bkc = (const float*)d_in[8];
  const float* Wg  = (const float*)d_in[9];
  const float* bg  = (const float*)d_in[10];
  const float* Wb  = (const float*)d_in[11];
  const float* bb  = (const float*)d_in[12];

  char* ws = (char*)d_ws;
  auto alloc = [&](size_t bytes) -> char* {
    char* p = ws;
    ws += (bytes + 255) & ~size_t(255);
    return p;
  };
  unsigned short* xb   = (unsigned short*)alloc((size_t)SEQ * Hdim * 2);
  unsigned short* wqb  = (unsigned short*)alloc((size_t)Hdim * Hdim * 2);
  unsigned short* wkb  = (unsigned short*)alloc((size_t)Hdim * Hdim * 2);
  unsigned short* wvb  = (unsigned short*)alloc((size_t)VD * Hdim * 2);
  unsigned short* wob  = (unsigned short*)alloc((size_t)Hdim * VD * 2);
  unsigned short* qraw = (unsigned short*)alloc((size_t)SEQ * Hdim * 2);
  unsigned short* kraw = (unsigned short*)alloc((size_t)SEQ * Hdim * 2);
  unsigned short* vbuf = (unsigned short*)alloc((size_t)SEQ * VD * 2);
  unsigned short* wqcb = (unsigned short*)alloc((size_t)CKW * Hdim * KDIM * 2);
  unsigned short* wkcb = (unsigned short*)alloc((size_t)CKW * Hdim * KDIM * 2);
  unsigned short* qcv  = (unsigned short*)alloc((size_t)SEQ * Hdim * 2);
  unsigned short* kcv  = (unsigned short*)alloc((size_t)SEQ * Hdim * 2);
  unsigned short* attn = (unsigned short*)alloc((size_t)SEQ * VD * 2);
  float* gateb = (float*)alloc((size_t)SEQ * HK * 4);
  float* betab = (float*)alloc((size_t)SEQ * HK * 4);

  auto cvt = [&](const float* s, unsigned short* d, int n) {
    cvt_f32_bf16<<<(n + 255) / 256, 256, 0, stream>>>(s, d, n);
  };
  cvt(x, xb, SEQ * Hdim);
  cvt(Wq, wqb, Hdim * Hdim);
  cvt(Wk, wkb, Hdim * Hdim);
  cvt(Wv, wvb, VD * Hdim);
  cvt(Wo, wob, Hdim * VD);
  repack_conv<<<(Hdim * KDIM * CKW + 255) / 256, 256, 0, stream>>>(
      Wqc, wqcb, Hdim * KDIM * CKW);
  repack_conv<<<(Hdim * KDIM * CKW + 255) / 256, 256, 0, stream>>>(
      Wkc, wkcb, Hdim * KDIM * CKW);

  // projections (bf16 out)
  gemm_nt<true><<<dim3(Hdim / 128, SEQ / 128), 256, 0, stream>>>(
      xb, wqb, qraw, SEQ, Hdim, Hdim);
  gemm_nt<true><<<dim3(Hdim / 128, SEQ / 128), 256, 0, stream>>>(
      xb, wkb, kraw, SEQ, Hdim, Hdim);
  gemm_nt<true><<<dim3(VD / 128, SEQ / 128), 256, 0, stream>>>(
      xb, wvb, vbuf, SEQ, VD, Hdim);

  // grouped causal conv + silu
  conv_silu<<<dim3(Hdim / 16, SEQ / 128), 256, 0, stream>>>(qraw, wqcb, bqc, qcv);
  conv_silu<<<dim3(Hdim / 16, SEQ / 128), 256, 0, stream>>>(kraw, wkcb, bkc, kcv);

  // gates
  gate_beta_kernel<<<SEQ, 256, 0, stream>>>(x, Wg, bg, Wb, bb, gateb, betab);

  // chunked scan: 32 workgroups = (head, half-VEFF); ~94.4 KB dynamic LDS
  float* Sf = (float*)d_out + (size_t)SEQ * Hdim;
  size_t smem = (size_t)KDIM * SSTR * 4 + 3 * (CH * VSTR * 2) + CH * ASTR * 2 +
                3 * CH * 4;
  scan_kernel<<<32, 256, smem, stream>>>(qcv, kcv, vbuf, gateb, betab, attn, Sf);

  // output projection (f32 out, directly into d_out head)
  gemm_nt<false><<<dim3(Hdim / 128, SEQ / 128), 256, 0, stream>>>(
      attn, wob, (float*)d_out, SEQ, Hdim, VD);
}
